// Int4Linear_72662256714090
// MI455X (gfx1250) — matrix operand order repaired and checked by reading, not compiled, not run on hardware
//
#include <hip/hip_runtime.h>

#define K_DIM 4096
#define N_DIM 4096
#define BM 128
#define BN 128
#define BK 64
#define ASTR 72            // padded LDS row stride in bf16 elements (144 B)

typedef __attribute__((ext_vector_type(16))) __bf16 v16bf;
typedef __attribute__((ext_vector_type(8)))  __bf16 v8bf;
typedef __attribute__((ext_vector_type(4)))  __bf16 v4bf;
typedef __attribute__((ext_vector_type(8)))  float  v8f;

union Frag {
    v16bf f;
    v8bf  h[2];
};

__global__ __launch_bounds__(256, 2)
void Int4Linear_wmma_kernel(const float* __restrict__ x,
                            const int*   __restrict__ qw,
                            const float* __restrict__ scale,
                            const int*   __restrict__ zp,
                            const float* __restrict__ bias,
                            float* __restrict__ out)
{
    __shared__ __bf16 sA[2][BM * ASTR];
    __shared__ __bf16 sB[2][BM * ASTR];

    const int tid = threadIdx.x;
    const int m0  = blockIdx.y * BM;
    const int n0  = blockIdx.x * BN;

    // ---- staging map: thread -> 8 rows (stride 16) x one float4 K-chunk ----
    const int srow = tid >> 4;          // 0..15
    const int scol = (tid & 15) << 2;   // k offset within tile: 0,4,...,60

    // loop-invariant 32-bit byte offsets; k advance goes into the scalar base,
    // the i*16-row step folds into the 24-bit instruction immediate.
    const unsigned ro_x = ((unsigned)(m0 + srow) * K_DIM + (unsigned)scol) * 4u;
    const unsigned ro_w = ((unsigned)(n0 + srow) * K_DIM + (unsigned)scol) * 4u;
    const unsigned RSTEP = 16u * K_DIM * 4u;   // 262144 B, fits signed-24 imm

    // per-output-channel dequant params:  w = q*scale + (-zp*scale)
    float psc[8], nzs[8];
#pragma unroll
    for (int i = 0; i < 8; ++i) {
        const int n = n0 + srow + 16 * i;
        psc[i] = scale[n];
        nzs[i] = -(float)zp[n] * psc[i];
    }

    float4 ra[8];
    int4   rb[8];

    auto gload = [&](int kb) {
        const char* xk = (const char*)x  + (size_t)kb * 4;
        const char* wk = (const char*)qw + (size_t)kb * 4;
#pragma unroll
        for (int i = 0; i < 8; ++i) {
            ra[i] = *(const float4*)(xk + (ro_x + i * RSTEP));
            rb[i] = *(const int4*)  (wk + (ro_w + i * RSTEP));
        }
    };

    auto cvtstore = [&](int buf) {
#pragma unroll
        for (int i = 0; i < 8; ++i) {
            const int r = srow + 16 * i;
            v4bf va;
            va.x = (__bf16)ra[i].x;  va.y = (__bf16)ra[i].y;
            va.z = (__bf16)ra[i].z;  va.w = (__bf16)ra[i].w;
            *(v4bf*)(&sA[buf][r * ASTR + scol]) = va;

            v4bf vb;
            vb.x = (__bf16)fmaf((float)rb[i].x, psc[i], nzs[i]);
            vb.y = (__bf16)fmaf((float)rb[i].y, psc[i], nzs[i]);
            vb.z = (__bf16)fmaf((float)rb[i].z, psc[i], nzs[i]);
            vb.w = (__bf16)fmaf((float)rb[i].w, psc[i], nzs[i]);
            *(v4bf*)(&sB[buf][r * ASTR + scol]) = vb;
        }
    };

    // ---- wave/lane layout for compute: 4x2 wave grid, 32x64 per wave ----
    const int lane = tid & 31;
    const int wave = tid >> 5;
    const int wm   = wave & 3;   // 0..3 along M
    const int wn   = wave >> 2;  // 0..1 along N
    const int lo   = lane & 15;
    const int hi   = lane >> 4;  // 0 or 1 (lane-half)

    v8f acc[2][4] = {};

    auto compute = [&](int buf) {
        // Batch ALL fragment loads for the BK tile first, then one WMMA burst:
        // waits become progressive dscnt thresholds instead of dscnt==0 stalls.
        Frag a[2][2];
        Frag b[2][4];
#pragma unroll
        for (int kx = 0; kx < 2; ++kx) {
            const int kk = kx * 32;
#pragma unroll
            for (int mt = 0; mt < 2; ++mt) {
                // A 16x32 bf16: lanes 0-15 K={0..7,16..23}, lanes 16-31 K={8..15,24..31}
                const __bf16* p = &sA[buf][(wm * 32 + mt * 16 + lo) * ASTR + kk];
                a[kx][mt].h[0] = *(const v8bf*)(p + hi * 8);
                a[kx][mt].h[1] = *(const v8bf*)(p + 16 + hi * 8);
            }
#pragma unroll
            for (int nt = 0; nt < 4; ++nt) {
                // B 32x16 bf16: lanes 0-15 K={0..7,16..23}, lanes 16-31 K={8..15,24..31}
                const __bf16* p = &sB[buf][(wn * 64 + nt * 16 + lo) * ASTR + kk];
                b[kx][nt].h[0] = *(const v8bf*)(p + hi * 8);
                b[kx][nt].h[1] = *(const v8bf*)(p + 16 + hi * 8);
            }
        }
#pragma unroll
        for (int kx = 0; kx < 2; ++kx) {
#pragma unroll
            for (int nt = 0; nt < 4; ++nt) {
#pragma unroll
                for (int mt = 0; mt < 2; ++mt) {
                    acc[mt][nt] = __builtin_amdgcn_wmma_f32_16x16x32_bf16(
                        false, a[kx][mt].f, false, b[kx][nt].f,
                        (short)0, acc[mt][nt], false, false);
                }
            }
        }
    };

    // ---- software-pipelined K loop (double-buffered LDS) ----
    gload(0);
    cvtstore(0);
    __syncthreads();

    const int KT = K_DIM / BK;  // 64
    for (int kt = 0; kt < KT; ++kt) {
        const int cur = kt & 1;
        if (kt + 1 < KT) gload((kt + 1) * BK);   // issue next-tile loads first
        compute(cur);                            // WMMAs hide VMEM latency
        if (kt + 1 < KT) cvtstore(cur ^ 1);      // dequant/cvt into other buffer
        __syncthreads();
    }

    // ---- epilogue: bias add + store (C/D layout: VGPR j -> M=j/j+8, lane -> N) ----
#pragma unroll
    for (int nt = 0; nt < 4; ++nt) {
        const int n  = n0 + wn * 64 + nt * 16 + lo;
        const float bv = bias[n];
#pragma unroll
        for (int mt = 0; mt < 2; ++mt) {
            const int mbase = m0 + wm * 32 + mt * 16 + hi * 8;
#pragma unroll
            for (int j = 0; j < 8; ++j) {
                out[(size_t)(mbase + j) * N_DIM + n] = acc[mt][nt][j] + bv;
            }
        }
    }
}

extern "C" void kernel_launch(void* const* d_in, const int* in_sizes, int n_in,
                              void* d_out, int out_size, void* d_ws, size_t ws_size,
                              hipStream_t stream) {
    const float* x    = (const float*)d_in[0];
    const int*   qw   = (const int*)d_in[1];
    const float* sc   = (const float*)d_in[2];
    const int*   zp   = (const int*)d_in[3];
    const float* bias = (const float*)d_in[4];
    float*       out  = (float*)d_out;

    const int M = in_sizes[0] / K_DIM;   // 2*2048 = 4096
    dim3 grid(N_DIM / BN, M / BM);
    dim3 block(256);
    Int4Linear_wmma_kernel<<<grid, block, 0, stream>>>(x, qw, sc, zp, bias, out);
}
